// Non_Local_10015863734430
// MI455X (gfx1250) — matrix operand order repaired and checked by reading, not compiled, not run on hardware
//
#include <hip/hip_runtime.h>
#include <hip/hip_bf16.h>

// ---------------------------------------------------------------------------
// Non-local attention block (flash-attention style) for MI455X / gfx1250.
//   B=4, C=128, CH=64, N=64*64=4096.
// Phase 1: 1x1-conv projections (fp32 VALU) -> f16 q/k_masked/v in workspace.
// Phase 2: per-16-query wave, online-softmax attention over 64-key blocks with
//          v_wmma_f32_16x16x32_f16; row reductions done with VALU DPP
//          (no ds_bpermute), P transposed through a padded LDS tile.
// ---------------------------------------------------------------------------

typedef _Float16 h16;
typedef __attribute__((ext_vector_type(8)))  _Float16 v8h;
typedef __attribute__((ext_vector_type(16))) _Float16 v16h;
typedef __attribute__((ext_vector_type(8)))  float    v8f;

#define CDIM   128
#define CHDIM  64
#define NPIX   4096
#define BATCH  4
#define PLD    72   // P-tile row stride in halves (144B: 16B-aligned, bank-clean)

// ---- WMMA helper ----------------------------------------------------------

__device__ __forceinline__ v8f wmma_f16(v16h a, v16h b, v8f c) {
  // D = A(16x32 f16) * B(32x16 f16) + C(16x16 f32)
  return __builtin_amdgcn_wmma_f32_16x16x32_f16(
      false, a, false, b, (short)0, c, false, false);
}

// A-fragment (16x32, f16), row-major source [16][ld] (ld in halves).
// lanes 0-15: row m=lane, K=0..15; lanes 16-31: K=16..31.
__device__ __forceinline__ v16h load_a16(const h16* p, int ld, int m, int akb) {
  v8h lo = *(const v8h*)(p + m * ld + akb);
  v8h hi = *(const v8h*)(p + m * ld + akb + 8);
  return __builtin_shufflevector(lo, hi,
      0, 1, 2, 3, 4, 5, 6, 7, 8, 9, 10, 11, 12, 13, 14, 15);
}

// ---- VALU-only 16-lane row reductions via DPP -----------------------------
// quad_perm xor1 (0xB1), quad_perm xor2 (0x4E), row_half_mirror (0x141),
// row_mirror (0x140): classic butterfly, stays inside each 16-lane row.

template <int CTRL>
__device__ __forceinline__ float dpp_bcast(float v) {
  int i = __builtin_amdgcn_update_dpp(0, __float_as_int(v), CTRL, 0xf, 0xf,
                                      true);
  return __int_as_float(i);
}

__device__ __forceinline__ float rowmax16(float v) {
  v = fmaxf(v, dpp_bcast<0xB1>(v));
  v = fmaxf(v, dpp_bcast<0x4E>(v));
  v = fmaxf(v, dpp_bcast<0x141>(v));
  v = fmaxf(v, dpp_bcast<0x140>(v));
  return v;
}

__device__ __forceinline__ float rowsum16(float v) {
  v += dpp_bcast<0xB1>(v);
  v += dpp_bcast<0x4E>(v);
  v += dpp_bcast<0x141>(v);
  v += dpp_bcast<0x140>(v);
  return v;
}

// ---------------------------------------------------------------------------
// Phase 1: projections. grid = BATCH * (NPIX/64) blocks, 256 threads.
// ---------------------------------------------------------------------------
__global__ __launch_bounds__(256)
void nl_proj_kernel(const float* __restrict__ x, const float* __restrict__ mask,
                    const float* __restrict__ Wq, const float* __restrict__ bq,
                    const float* __restrict__ Wk, const float* __restrict__ bk,
                    const float* __restrict__ Wv, const float* __restrict__ bv,
                    h16* __restrict__ qh, h16* __restrict__ kh,
                    h16* __restrict__ vh) {
  __shared__ float xs[CDIM * 64];  // 32 KB tile: x[b][0..127][p0..p0+63]

  const int b  = blockIdx.x >> 6;
  const int p0 = (blockIdx.x & 63) << 6;
  const float* xb = x + (size_t)b * CDIM * NPIX + p0;

  for (int l = threadIdx.x; l < CDIM * 64; l += 256) {
    const int c = l >> 6, p = l & 63;
    xs[l] = xb[(size_t)c * NPIX + p];
  }
  __syncthreads();

  const int d = threadIdx.x;
  const float* wrow;
  if (d < 64)       wrow = Wq + d * CDIM;
  else if (d < 128) wrow = Wk + (d - 64) * CDIM;
  else              wrow = Wv + (d - 128) * CDIM;

  float acc[64];
#pragma unroll
  for (int p = 0; p < 64; ++p) acc[p] = 0.0f;

  for (int c = 0; c < CDIM; ++c) {
    const float w = wrow[c];
    const float* xr = xs + c * 64;
#pragma unroll
    for (int p = 0; p < 64; ++p) acc[p] = fmaf(w, xr[p], acc[p]);
  }

  if (d < 64) {                       // q: layout [b][pixel][ch]
    const float bias = bq[d];
#pragma unroll 8
    for (int p = 0; p < 64; ++p)
      qh[((size_t)(b * NPIX + p0 + p)) * CHDIM + d] = (h16)(acc[p] + bias);
  } else if (d < 128) {               // k_masked: layout [b][pixel][ch]
    const int dd = d - 64;
    const float bias = bk[dd];
#pragma unroll 8
    for (int p = 0; p < 64; ++p) {
      const float km =
          (acc[p] + bias) * (1.0f - mask[(size_t)b * NPIX + p0 + p]);
      kh[((size_t)(b * NPIX + p0 + p)) * CHDIM + dd] = (h16)km;
    }
  } else {                            // v: layout [b][ch][pixel]
    const int dd = d - 128;
    const float bias = bv[dd];
    h16* vrow = vh + ((size_t)(b * CDIM + dd)) * NPIX + p0;
#pragma unroll 8
    for (int p = 0; p < 64; ++p) vrow[p] = (h16)(acc[p] + bias);
  }
}

// ---------------------------------------------------------------------------
// Phase 2: flash attention. One wave = 16 queries x all 4096 keys,
// 64 keys per iteration. grid = 128 blocks x 256 threads (1024 waves).
// ---------------------------------------------------------------------------
__global__ __launch_bounds__(256)
void nl_attn_kernel(const h16* __restrict__ qh, const h16* __restrict__ kh,
                    const h16* __restrict__ vh, const float* __restrict__ mask,
                    const float* __restrict__ x, const float* __restrict__ gamma,
                    float* __restrict__ out) {
  __shared__ __align__(128) h16 plds[8 * 16 * PLD];  // per-wave P tile (16x64)

  const int lane = threadIdx.x & 31;
  const int widx = threadIdx.x >> 5;
  const int gw   = blockIdx.x * 8 + widx;
  const int b    = gw >> 8;           // 256 query tiles per batch
  const int i0   = (gw & 255) << 4;   // query base (16 rows)

  const int m   = lane & 15;          // A-frag row / B-frag column
  const int hi  = lane >> 4;          // half-wave select
  const int akb = hi << 4;            // A-frag K base: 0 or 16
  const int bko = hi << 4;            // B-frag K base: 0 or 16

  h16* pt = plds + widx * 16 * PLD;

  // Resident Q fragments (queries i0..i0+15, channels 0..63).
  const h16* qbase = qh + ((size_t)(b * NPIX + i0)) * CHDIM;
  const v16h qa0 = load_a16(qbase,      CHDIM, m, akb);  // ch  0..31
  const v16h qa1 = load_a16(qbase + 32, CHDIM, m, akb);  // ch 32..63

  float mrow[8], lrow[8];
#pragma unroll
  for (int r = 0; r < 8; ++r) { mrow[r] = -3.0e38f; lrow[r] = 0.0f; }
  v8f oacc[8];
#pragma unroll
  for (int t = 0; t < 8; ++t) {
    v8f z = {};
    oacc[t] = z;
  }

  const h16* kbase = kh + ((size_t)(b * NPIX + m)) * CHDIM + bko;

#pragma unroll 1
  for (int j0 = 0; j0 < NPIX; j0 += 64) {
    // ---- S = Q * K^T for 64 keys (four 16-key column tiles) -------------
    v8f s[4];
#pragma unroll
    for (int u = 0; u < 4; ++u) {
      const h16* kr = kbase + (size_t)(j0 + u * 16) * CHDIM;
      const v16h klo = *(const v16h*)(kr);        // channels  0..31
      const v16h khi = *(const v16h*)(kr + 32);   // channels 32..63
      v8f z = {};
      z = wmma_f16(qa0, klo, z);
      s[u] = wmma_f16(qa1, khi, z);
    }

    // ---- online softmax over the 64 key columns (DPP reductions) --------
    float corr[8];
#pragma unroll
    for (int r = 0; r < 8; ++r) {
      float mx = fmaxf(fmaxf(s[0][r], s[1][r]), fmaxf(s[2][r], s[3][r]));
      mx = rowmax16(mx);
      const float mn = fmaxf(mrow[r], mx);
      const float cr = __expf(mrow[r] - mn);
      const float p0 = __expf(s[0][r] - mn);
      const float p1 = __expf(s[1][r] - mn);
      const float p2 = __expf(s[2][r] - mn);
      const float p3 = __expf(s[3][r] - mn);
      float rs = (p0 + p1) + (p2 + p3);
      rs = rowsum16(rs);
      lrow[r] = lrow[r] * cr + rs;
      mrow[r] = mn;
      corr[r] = cr;
      h16* prow = pt + (r + 8 * hi) * PLD + m;    // P tile [row][col]
      prow[0]  = (h16)p0;
      prow[16] = (h16)p1;
      prow[32] = (h16)p2;
      prow[48] = (h16)p3;
    }

    // Rescale existing output accumulators (same row map as S tiles).
#pragma unroll
    for (int t = 0; t < 8; ++t)
#pragma unroll
      for (int r = 0; r < 8; ++r) oacc[t][r] *= corr[r];

    // P tile is wave-private; DS ops are in-order per wave — drain stores.
    asm volatile("s_wait_dscnt 0" ::: "memory");
    const v16h pa0 = load_a16(pt,      PLD, m, akb);  // P keys j0   ..j0+31
    const v16h pa1 = load_a16(pt + 32, PLD, m, akb);  // P keys j0+32..j0+63

    // ---- out += P * V^T (8 channel tiles of 16) -------------------------
#pragma unroll
    for (int t = 0; t < 8; ++t) {
      const h16* vrow =
          vh + ((size_t)(b * CDIM + t * 16 + m)) * NPIX + j0 + bko;
      const v16h vb0 = *(const v16h*)vrow;
      const v16h vb1 = *(const v16h*)(vrow + 32);
      oacc[t] = wmma_f16(pa0, vb0, oacc[t]);
      oacc[t] = wmma_f16(pa1, vb1, oacc[t]);
    }
  }

  // ---- epilogue: normalize, mask, gamma, residual, store ----------------
  const float gm = gamma[0];
  float inv[8], mk[8];
#pragma unroll
  for (int r = 0; r < 8; ++r) {
    const int i = i0 + r + 8 * hi;
    inv[r] = 1.0f / lrow[r];
    mk[r]  = mask[(size_t)b * NPIX + i] * gm;
  }
#pragma unroll
  for (int t = 0; t < 8; ++t) {
    const int c = t * 16 + m;
    const float* xrow = x   + ((size_t)(b * CDIM + c)) * NPIX;
    float*       orow = out + ((size_t)(b * CDIM + c)) * NPIX;
#pragma unroll
    for (int r = 0; r < 8; ++r) {
      const int i = i0 + r + 8 * hi;
      orow[i] = (oacc[t][r] * inv[r]) * mk[r] + xrow[i];
    }
  }
}

// ---------------------------------------------------------------------------
extern "C" void kernel_launch(void* const* d_in, const int* in_sizes, int n_in,
                              void* d_out, int out_size, void* d_ws,
                              size_t ws_size, hipStream_t stream) {
  (void)in_sizes; (void)n_in; (void)out_size; (void)ws_size;
  const float* x     = (const float*)d_in[0];
  const float* mask  = (const float*)d_in[1];
  const float* Wq    = (const float*)d_in[2];
  const float* bq    = (const float*)d_in[3];
  const float* Wk    = (const float*)d_in[4];
  const float* bk    = (const float*)d_in[5];
  const float* Wv    = (const float*)d_in[6];
  const float* bv    = (const float*)d_in[7];
  const float* gamma = (const float*)d_in[8];
  float* out = (float*)d_out;

  // Workspace: q (2 MB) | k_masked (2 MB) | v (4 MB), all f16.
  h16* qh = (h16*)d_ws;
  h16* kh = qh + (size_t)BATCH * NPIX * CHDIM;
  h16* vh = kh + (size_t)BATCH * NPIX * CHDIM;

  nl_proj_kernel<<<BATCH * (NPIX / 64), 256, 0, stream>>>(
      x, mask, Wq, bq, Wk, bk, Wv, bv, qh, kh, vh);

  nl_attn_kernel<<<(BATCH * (NPIX / 16)) / 8, 256, 0, stream>>>(
      qh, kh, vh, mask, x, gamma, out);
}
